// VecLocal2d_54709293416450
// MI455X (gfx1250) — matrix operand order repaired and checked by reading, not compiled, not run on hardware
//
#include <hip/hip_runtime.h>

// ---------- types ----------
typedef __attribute__((ext_vector_type(16))) __bf16 v16bf;
typedef __attribute__((ext_vector_type(8)))  __bf16 v8bf;
typedef __attribute__((ext_vector_type(4)))  __bf16 v4bf;
typedef __attribute__((ext_vector_type(8)))  float  v8f;
typedef __attribute__((ext_vector_type(4)))  float  v4f;

// ---------- problem constants ----------
constexpr int BATCH   = 160;      // B * CAT
constexpr int C_IN    = 64;
constexpr int HW      = 32;       // H = W = Ho = Wo
constexpr int C_OUT   = 128;
constexpr int KDIM    = 576;      // C_IN * 3 * 3
constexpr int W_OSTR  = 1024 * KDIM;   // weight stride per output channel (589824)
constexpr int OUT_NSTR= C_OUT * 1024;  // output stride per batch row (131072)

// ---------- tiling ----------
constexpr int M_BLK    = 80;           // batch rows per workgroup (2 WGs per location)
constexpr int M_TILES  = M_BLK / 16;   // 5
constexpr int KC       = 64;           // K chunk staged for B
constexpr int NCHUNK   = KDIM / KC;    // 9
constexpr int A_STRIDE = KDIM + 8;     // 584 halves -> 1168B: 16B aligned, bank-conflict-free
constexpr int B_STRIDE = KC + 8;       // 72 halves  -> 144B : 16B aligned, bank-conflict-free

constexpr int LDS_HALVES = M_BLK * A_STRIDE + C_OUT * B_STRIDE; // 46720 + 9216
// dynamic LDS bytes = 111,872 (2 workgroups fit in 320KB WGP LDS)

__global__ void __launch_bounds__(256)
local2d_wmma_bf16(const float* __restrict__ x,     // (160, 64, 32, 32)
                  const float* __restrict__ w,     // (128, 32, 32, 576)
                  const float* __restrict__ bias,  // (128, 32, 32)
                  float* __restrict__ out)         // (160, 128, 32, 32)
{
    extern __shared__ __bf16 lds[];
    __bf16* Alds = lds;                          // [M_BLK][A_STRIDE]
    __bf16* Blds = lds + M_BLK * A_STRIDE;       // [C_OUT][B_STRIDE]

    const int bid  = blockIdx.x;
    const int loc  = bid >> 1;                   // 0..1023  (u*32 + v)
    const int u    = loc >> 5;
    const int v    = loc & 31;
    const int n0   = (bid & 1) * M_BLK;

    const int tid  = threadIdx.x;
    const int lane = tid & 31;
    const int wave = tid >> 5;                   // 0..7 -> output-channel column

    // ---------------- stage A (patches), full K, fp32 -> bf16 ----------------
    // triples (nn, c, ki): 80*64*3 = 15360 = 60 * 256
    for (int i = 0; i < 60; ++i) {
        int t  = tid + (i << 8);
        int ki = t % 3;
        int c  = (t / 3) & 63;
        int nn = t / 192;                        // 0..79
        int uu = u - 1 + ki;
        bool rowok = (unsigned)uu < 32u;
        const float* src = x + ((((n0 + nn) * 64 + c) * 32 + uu) * 32) + (v - 1);
        __bf16* dst = Alds + nn * A_STRIDE + c * 9 + ki * 3;
        #pragma unroll
        for (int j = 0; j < 3; ++j) {
            int vv = v - 1 + j;
            float f = (rowok && ((unsigned)vv < 32u)) ? src[j] : 0.0f;
            dst[j] = (__bf16)f;
        }
    }

    // ---------------- accumulators preloaded with bias ----------------
    const int oc      = (wave << 4) + (lane & 15);   // this lane's output channel
    const int halfsel = lane >> 4;                   // 0 / 1
    const float bv = bias[oc * 1024 + loc];
    v8f acc[M_TILES];
    #pragma unroll
    for (int m = 0; m < M_TILES; ++m) {
        #pragma unroll
        for (int r = 0; r < 8; ++r) acc[m][r] = bv;
    }

    // fragment base pointers (LDS)
    const __bf16* bfrag = Blds + oc * B_STRIDE + (halfsel << 3);        // +8 halves for hi lanes
    const __bf16* afrag = Alds + (lane & 15) * A_STRIDE + (halfsel << 3); // +8 halves for hi lanes

    // ---------------- K loop: stage B chunk, 2 WMMA k-steps ----------------
    const float* wloc = w + loc * KDIM;
    for (int ch = 0; ch < NCHUNK; ++ch) {
        __syncthreads();
        // stage weights: 128 channels x 64 floats = 2048 float4 = 8 per thread
        const float* wsrc = wloc + ch * KC;
        #pragma unroll
        for (int q = 0; q < 8; ++q) {
            int idx = tid + (q << 8);            // 0..2047
            int o   = idx >> 4;                  // 0..127
            int f4  = idx & 15;                  // 0..15
            v4f d = *(const v4f*)(wsrc + (size_t)o * W_OSTR + f4 * 4);
            v4bf p = { (__bf16)d.x, (__bf16)d.y, (__bf16)d.z, (__bf16)d.w };
            *(v4bf*)(Blds + o * B_STRIDE + f4 * 4) = p;
        }
        __syncthreads();

        #pragma unroll
        for (int s = 0; s < 2; ++s) {
            const int kk = ch * KC + s * 32;     // global k of this k-step
            // B fragment: segments K=[8h..8h+8) and K=[16+8h..+8)
            union { v16bf v; v8bf h[2]; } bu;
            bu.h[0] = *(const v8bf*)(bfrag + s * 32);
            bu.h[1] = *(const v8bf*)(bfrag + s * 32 + 16);
            #pragma unroll
            for (int m = 0; m < M_TILES; ++m) {
                // A fragment: segments K=[8h..8h+8) and K=[16+8h..+8)
                const __bf16* ap = afrag + (m * 16) * A_STRIDE + kk;
                union { v16bf v; v8bf h[2]; } au;
                au.h[0] = *(const v8bf*)(ap);
                au.h[1] = *(const v8bf*)(ap + 16);
                acc[m] = __builtin_amdgcn_wmma_f32_16x16x32_bf16(
                    false, au.v, false, bu.v, (short)0, acc[m], false, false);
            }
        }
    }

    // ---------------- write out (+bias already folded in) ----------------
    float* obase = out + (size_t)oc * 1024 + loc;
    #pragma unroll
    for (int m = 0; m < M_TILES; ++m) {
        int nb = n0 + m * 16 + (halfsel << 3);
        #pragma unroll
        for (int r = 0; r < 8; ++r) {
            obase[(size_t)(nb + r) * OUT_NSTR] = acc[m][r];
        }
    }
}

extern "C" void kernel_launch(void* const* d_in, const int* in_sizes, int n_in,
                              void* d_out, int out_size, void* d_ws, size_t ws_size,
                              hipStream_t stream) {
    const float* x  = (const float*)d_in[0];
    const float* w  = (const float*)d_in[1];
    const float* b  = (const float*)d_in[2];
    float* out      = (float*)d_out;

    const size_t shmem = (size_t)LDS_HALVES * sizeof(__bf16);   // 111,872 bytes
    hipFuncSetAttribute((const void*)local2d_wmma_bf16,
                        hipFuncAttributeMaxDynamicSharedMemorySize, (int)shmem);

    dim3 grid(1024 * 2);   // one block per (location, M-half)
    dim3 block(256);       // 8 waves
    local2d_wmma_bf16<<<grid, block, shmem, stream>>>(x, w, b, out);
}
